// Involution3D_43404939493776
// MI455X (gfx1250) — matrix operand order repaired and checked by reading, not compiled, not run on hardware
//
#include <hip/hip_runtime.h>
#include <hip/hip_bf16.h>

// ---------------------------------------------------------------------------
// Involution3D for MI455X (gfx1250, wave32, WMMA).
//   K1: r = x @ w_reduce + b  (bf16 WMMA, fp32 acc) + partial GN stats
//   K2: finalize GroupNorm mean / rsqrt(var+eps)  (8 scalars)
//   K3: gn+relu -> weight = A @ w_span + b (bf16 WMMA) -> 27-tap involution
// x/w loads and out stores are 128-bit, r traffic is 64-bit; output stores are non-temporal to keep
// x (67MB) + r (33MB) resident in the 192MB L2 for the gather phase.
// ---------------------------------------------------------------------------

typedef __attribute__((ext_vector_type(16))) __bf16 v16bf;
typedef __attribute__((ext_vector_type(8)))  __bf16 v8bf;
typedef __attribute__((ext_vector_type(4)))  __bf16 v4bf;
typedef __attribute__((ext_vector_type(8)))  float  v8f;
typedef __attribute__((ext_vector_type(4)))  float  v4f;

#define B_      2
#define D_      32
#define H_      64
#define W_      64
#define C_      64
#define S_      (D_*H_*W_)    // 131072 spatial per batch
#define N_      (B_*S_)       // 262144 positions
#define GROUPS_ 4
#define CPG_    16
#define NSPAN   108           // 27*4
#define NSPAN_P 112           // padded to 7 WMMA n-tiles
#define GN_EPS  1e-3f

// --- helpers ----------------------------------------------------------------

__device__ inline v4bf cvt4(v4f v) {
  v4bf r;
  r.x = (__bf16)v.x; r.y = (__bf16)v.y;
  r.z = (__bf16)v.z; r.w = (__bf16)v.w;
  return r;
}

// A (16x32 bf16): lane&15 = M row; lanes<16 hold K {0..7,16..23},
// lanes>=16 hold K {8..15,24..31}. base points at (row, khalf*32).
__device__ inline v16bf load_a_frag(const __bf16* base, int lane) {
  const int kb = (lane >> 4) << 3;
  union { v16bf v; v8bf h[2]; } u;
  u.h[0] = *(const v8bf*)(base + kb);        // K = kb .. kb+7
  u.h[1] = *(const v8bf*)(base + 16 + kb);   // K = 16+kb .. 16+kb+7
  return u.v;
}

// B (32x16 bf16): lane&15 = N col; lanes<16 hold K {0..7,16..23},
// lanes>=16 hold K {8..15,24..31}.
__device__ inline v16bf load_b_frag(const __bf16* lds, int stride, int k0,
                                    int n0, int lane) {
  const __bf16* p = lds + (k0 + ((lane >> 4) << 3)) * stride + n0 + (lane & 15);
  v16bf v;
#pragma unroll
  for (int s = 0; s < 8; ++s) {
    v[s]     = p[s * stride];
    v[8 + s] = p[(16 + s) * stride];
  }
  return v;
}

__device__ inline v8f wmma_bf16(v16bf a, v16bf b, v8f c) {
  return __builtin_amdgcn_wmma_f32_16x16x32_bf16(
      /*neg_a=*/false, a, /*neg_b=*/false, b,
      /*c_mod=*/(short)0, c, /*reuse_a=*/false, /*reuse_b=*/false);
}

// --- Kernel 1: reduce GEMM + GN partial stats -------------------------------
__global__ __launch_bounds__(256) void k_reduce_gemm(
    const float* __restrict__ x, const float* __restrict__ w_reduce,
    const float* __restrict__ b_reduce, __bf16* __restrict__ r_bf,
    float* __restrict__ stats) {
  __shared__ __align__(16) __bf16 lds_x[128 * 64];     // 16 KB
  __shared__ __align__(16) __bf16 lds_w[64 * 64];      //  8 KB
  __shared__ __align__(16) float  lds_stage[128 * 64]; // 32 KB

  const int  tid  = threadIdx.x;
  const int  lane = tid & 31;
  const int  wave = tid >> 5;
  const long row0 = (long)blockIdx.x * 128;

  const v4f* xq = (const v4f*)x;             // 16 quads per row
  const v4f* wq = (const v4f*)w_reduce;

#pragma unroll
  for (int e = 0; e < 4; ++e) {              // stage w_reduce (64x64) b128
    const int qi = e * 256 + tid;            // 0..1023
    *(v4bf*)&lds_w[qi * 4] = cvt4(wq[qi]);
  }
#pragma unroll
  for (int e = 0; e < 8; ++e) {              // stage 128x64 x tile, b128
    const int qi = e * 256 + tid;            // 0..2047
    *(v4bf*)&lds_x[qi * 4] = cvt4(xq[row0 * 16 + qi]);
  }
  __syncthreads();

  const __bf16* arow = &lds_x[(wave * 16 + (lane & 15)) * 64];
  const v16bf a0 = load_a_frag(arow, lane);
  const v16bf a1 = load_a_frag(arow + 32, lane);

  const int batch = (int)(row0 / S_);        // tiles never cross batches

  for (int t = 0; t < 4; ++t) {              // n-tile == channel group
    const v16bf b0 = load_b_frag(lds_w, 64, 0,  16 * t, lane);
    const v16bf b1 = load_b_frag(lds_w, 64, 32, 16 * t, lane);
    v8f acc = {};
    acc = wmma_bf16(a0, b0, acc);
    acc = wmma_bf16(a1, b1, acc);

    const float bias = b_reduce[16 * t + (lane & 15)];
    float s = 0.f, s2 = 0.f;
#pragma unroll
    for (int r = 0; r < 8; ++r) {
      const float v = acc[r] + bias;
      s  += v;
      s2 += v * v;
      const int lrow = wave * 16 + r + ((lane >> 4) << 3);
      lds_stage[lrow * 64 + 16 * t + (lane & 15)] = v;
    }
#pragma unroll
    for (int off = 16; off > 0; off >>= 1) { // wave32 tree reduce
      s  += __shfl_xor(s,  off);
      s2 += __shfl_xor(s2, off);
    }
    if (lane == 0) {
      atomicAdd(&stats[(batch * 4 + t) * 2 + 0], s);
      atomicAdd(&stats[(batch * 4 + t) * 2 + 1], s2);
    }
  }
  __syncthreads();

#pragma unroll
  for (int e = 0; e < 8; ++e) {              // packed bf16 quad writes (b64)
    const int qi = e * 256 + tid;            // 0..2047
    const v4f v = *(const v4f*)&lds_stage[qi * 4];
    *(v4bf*)&r_bf[row0 * 64 + qi * 4] = cvt4(v);
  }
}

// --- Kernel 2: finalize GroupNorm statistics --------------------------------
__global__ void k_stats(const float* __restrict__ stats,
                        float* __restrict__ stats2) {
  const int i = threadIdx.x;
  if (i < B_ * GROUPS_) {
    const float inv  = 1.0f / (float)((long)S_ * CPG_);
    const float mean = stats[i * 2 + 0] * inv;
    const float var  = stats[i * 2 + 1] * inv - mean * mean;
    stats2[i * 2 + 0] = mean;
    stats2[i * 2 + 1] = rsqrtf(var + GN_EPS);
  }
}

// --- Kernel 3: GN apply + span GEMM + involution ----------------------------
__global__ __launch_bounds__(256) void k_span_involution(
    const float* __restrict__ x, const __bf16* __restrict__ r_bf,
    const float* __restrict__ gamma, const float* __restrict__ beta,
    const float* __restrict__ w_span, const float* __restrict__ b_span,
    const float* __restrict__ stats2, float* __restrict__ out) {
  __shared__ __align__(16) __bf16 lds_a[64 * 64];        //  8 KB
  __shared__ __align__(16) __bf16 lds_ws[64 * NSPAN_P];  // 14 KB
  __shared__ __align__(16) float  lds_wt[64 * NSPAN_P];  // 28 KB

  const int  tid  = threadIdx.x;
  const int  lane = tid & 31;
  const int  wave = tid >> 5;
  const long row0 = (long)blockIdx.x * 64;   // one full W line: fixed b,d,h
  const int  batch = (int)(row0 / S_);
  const int  rem   = (int)(row0 - (long)batch * S_);
  const int  d = rem >> 12;                  // / (H*W)
  const int  h = (rem >> 6) & 63;

#pragma unroll
  for (int e = 0; e < 4; ++e) {              // GN + relu -> bf16 A tile (b64)
    const int qi = e * 256 + tid;            // 0..1023, quad of channels
    const int c = (qi * 4) & 63;             // 4 consecutive ch, same group
    const int g = c >> 4;
    const float mean = stats2[(batch * 4 + g) * 2 + 0];
    const float scal = stats2[(batch * 4 + g) * 2 + 1];
    const v4f  gm = *(const v4f*)&gamma[c];
    const v4f  bt = *(const v4f*)&beta[c];
    const v4bf rv = *(const v4bf*)&r_bf[row0 * 64 + qi * 4];
    v4f v;
    v.x = (float)rv.x; v.y = (float)rv.y;
    v.z = (float)rv.z; v.w = (float)rv.w;
    v = (v - mean) * scal * gm + bt;
    v.x = v.x > 0.f ? v.x : 0.f;  v.y = v.y > 0.f ? v.y : 0.f;
    v.z = v.z > 0.f ? v.z : 0.f;  v.w = v.w > 0.f ? v.w : 0.f;
    *(v4bf*)&lds_a[qi * 4] = cvt4(v);
  }
#pragma unroll
  for (int e = 0; e < 7; ++e) {              // stage w_span 64x108 (b128)
    const int qi = e * 256 + tid;            // 0..1791, quads (27 per row)
    if (qi < 64 * 27) {
      const int row = qi / 27;
      const int n0  = (qi - row * 27) * 4;   // row*108*4B is 16B-aligned
      const v4f v = *(const v4f*)&w_span[row * NSPAN + n0];
      *(v4bf*)&lds_ws[row * NSPAN_P + n0] = cvt4(v);
    }
  }
  {                                          // zero the 4 pad columns
    const int row = tid >> 2;
    lds_ws[row * NSPAN_P + NSPAN + (tid & 3)] = (__bf16)0.f;
  }
  __syncthreads();

  for (int job = wave; job < 4 * 7; job += 8) {  // 4 m-tiles x 7 n-tiles
    const int mt = job / 7, nt = job % 7;
    const __bf16* arow = &lds_a[(mt * 16 + (lane & 15)) * 64];
    const v16bf a0 = load_a_frag(arow, lane);
    const v16bf a1 = load_a_frag(arow + 32, lane);
    const v16bf b0 = load_b_frag(lds_ws, NSPAN_P, 0,  16 * nt, lane);
    const v16bf b1 = load_b_frag(lds_ws, NSPAN_P, 32, 16 * nt, lane);
    v8f acc = {};
    acc = wmma_bf16(a0, b0, acc);
    acc = wmma_bf16(a1, b1, acc);
    const int   n    = 16 * nt + (lane & 15);
    const float bias = (n < NSPAN) ? b_span[n] : 0.f;
#pragma unroll
    for (int r = 0; r < 8; ++r) {
      const int lrow = mt * 16 + r + ((lane >> 4) << 3);
      lds_wt[lrow * NSPAN_P + n] = acc[r] + bias;
    }
  }
  __syncthreads();

  // 27-tap involution; symmetric pad(1) == edge clamp. x is L2-resident.
#pragma unroll
  for (int q = 0; q < 4; ++q) {
    const int qi = q * 256 + tid;            // 0..1023 (64 rows x 16 quads)
    const int lr = qi >> 4;                  // local row == w coordinate
    const int c  = (qi & 15) << 2;
    const int g  = c >> 4;
    v4f acc = {0.f, 0.f, 0.f, 0.f};
#pragma unroll
    for (int i = 0; i < 3; ++i) {
      int dd = d + i - 1; dd = dd < 0 ? 0 : (dd >= D_ ? D_ - 1 : dd);
#pragma unroll
      for (int j = 0; j < 3; ++j) {
        int hh = h + j - 1; hh = hh < 0 ? 0 : (hh >= H_ ? H_ - 1 : hh);
        const long base = (((long)batch * D_ + dd) * H_ + hh) * W_ * 64;
#pragma unroll
        for (int l = 0; l < 3; ++l) {
          int ww = lr + l - 1; ww = ww < 0 ? 0 : (ww >= W_ ? W_ - 1 : ww);
          const int   k3  = (i * 3 + j) * 3 + l;
          const float wgt = lds_wt[lr * NSPAN_P + k3 * GROUPS_ + g];
          const v4f   xv  = *(const v4f*)&x[base + (long)ww * 64 + c];
          acc += xv * wgt;
        }
      }
    }
    // never re-read: bypass-cache store keeps x/r resident in L2
    __builtin_nontemporal_store(acc, (v4f*)&out[(row0 + lr) * 64 + c]);
  }
}

// --- Host launcher ----------------------------------------------------------
extern "C" void kernel_launch(void* const* d_in, const int* in_sizes, int n_in,
                              void* d_out, int out_size, void* d_ws,
                              size_t ws_size, hipStream_t stream) {
  const float* x        = (const float*)d_in[0];
  const float* w_reduce = (const float*)d_in[1];
  const float* b_reduce = (const float*)d_in[2];
  const float* gn_gamma = (const float*)d_in[3];
  const float* gn_beta  = (const float*)d_in[4];
  const float* w_span   = (const float*)d_in[5];
  const float* b_span   = (const float*)d_in[6];
  float* out = (float*)d_out;

  char*   ws     = (char*)d_ws;
  __bf16* r_bf   = (__bf16*)ws;                         // N*64 bf16 = 33.5 MB
  float*  stats  = (float*)(ws + (size_t)N_ * 64 * 2);  // 16 f32 (sum,sumsq)
  float*  stats2 = stats + 16;                          // 16 f32 (mean,rstd)

  (void)hipMemsetAsync(stats, 0, 16 * sizeof(float), stream);
  k_reduce_gemm<<<N_ / 128, 256, 0, stream>>>(x, w_reduce, b_reduce, r_bf,
                                              stats);
  k_stats<<<1, 32, 0, stream>>>(stats, stats2);
  k_span_involution<<<N_ / 64, 256, 0, stream>>>(x, r_bf, gn_gamma, gn_beta,
                                                 w_span, b_span, stats2, out);
}
